// Grapher_60533269070255
// MI455X (gfx1250) — matrix operand order repaired and checked by reading, not compiled, not run on hardware
//
#include <hip/hip_runtime.h>
#include <hip/hip_bf16.h>

// ---------------- problem constants ----------------
#define BB 4
#define CC 96
#define NN 3136          // 56*56
#define C2 192
#define KNN 9
#define NT_PER_B 196     // NN/16
#define NT_BLK 49        // NT_PER_B / 4 waves

typedef __attribute__((ext_vector_type(2)))  float  v2f;
typedef __attribute__((ext_vector_type(8)))  float  v8f;
typedef __attribute__((ext_vector_type(16))) __bf16 v16bf;

union BF16Frag { v16bf v; uint4 q[2]; };

#define DEVI static __device__ __forceinline__

DEVI v8f wmma_f32(v2f a, v2f b, v8f c) {
  // V_WMMA_F32_16X16X4_F32 : D = A(16x4) * B(4x16) + C
  return __builtin_amdgcn_wmma_f32_16x16x4_f32(false, a, false, b, (short)0, c, false, false);
}
DEVI v8f wmma_bf16(v16bf a, v16bf b, v8f c) {
  // V_WMMA_F32_16X16X32_BF16
  return __builtin_amdgcn_wmma_f32_16x16x32_bf16(false, a, false, b, (short)0, c, false, false);
}
DEVI void lds_wait() { __asm__ volatile("s_wait_dscnt 0" ::: "memory"); }

DEVI unsigned short f2bf(float f) {
  unsigned u = __float_as_uint(f);
  u += 0x7fffu + ((u >> 16) & 1u);
  return (unsigned short)(u >> 16);
}

// ---------------- K1: fc1 + bn1 -> xf[b][n][c] (f32) ----------------
__global__ __launch_bounds__(128) void k_fc1(
    const float* __restrict__ x, const float* __restrict__ w,
    const float* __restrict__ bias, const float* __restrict__ bn,
    float* __restrict__ xf) {
  const int wid  = threadIdx.x >> 5;
  const int lane = threadIdx.x & 31;
  const int half = lane >> 4, l16 = lane & 15;
  const int nt = (blockIdx.x % NT_BLK) * 4 + wid;
  const int b  = blockIdx.x / NT_BLK;
  const int n0 = nt * 16;
  const int n  = n0 + l16;
  const int koff = half ? 2 : 0;

  // A fragments: rows = positions n, K = input channel c (24 steps of K=4)
  v2f a[24];
  const float* xb = x + (size_t)b * CC * NN + n;
#pragma unroll
  for (int s = 0; s < 24; ++s) {
    int c = 4 * s + koff;
    a[s].x = xb[(size_t)c * NN];
    a[s].y = xb[(size_t)(c + 1) * NN];
  }
  for (int ot = 0; ot < 6; ++ot) {
    int o = ot * 16 + l16;
    v8f acc = {0.f,0.f,0.f,0.f,0.f,0.f,0.f,0.f};
#pragma unroll
    for (int s = 0; s < 24; ++s) {
      int c = 4 * s + koff;
      v2f bf; bf.x = w[o * CC + c]; bf.y = w[o * CC + c + 1];
      acc = wmma_f32(a[s], bf, acc);
    }
    float g = bn[o], be = bn[CC + o], mu = bn[2 * CC + o], va = bn[3 * CC + o];
    float inv = g / sqrtf(va + 1e-5f);
    float sh  = bias[o] * inv + be - mu * inv;
#pragma unroll
    for (int v = 0; v < 8; ++v) {
      int nr = n0 + v + half * 8;
      xf[((size_t)b * NN + nr) * CC + o] = acc[v] * inv + sh;
    }
  }
}

// ---------------- K1b: per-position L2 norm -> xn (bf16), sq (f32) ----------------
__global__ void k_norm(const float* __restrict__ xf,
                       unsigned short* __restrict__ xn, float* __restrict__ sq) {
  int p = blockIdx.x * blockDim.x + threadIdx.x;
  if (p >= BB * NN) return;
  const float4* v4 = (const float4*)(xf + (size_t)p * CC);
  float s = 0.f;
#pragma unroll 6
  for (int i = 0; i < 24; ++i) {
    float4 t = v4[i];
    s += t.x * t.x + t.y * t.y + t.z * t.z + t.w * t.w;
  }
  float r = 1.f / fmaxf(sqrtf(s), 1e-12f);
  sq[p] = (s * r) * r;
  uint2* o = (uint2*)(xn + (size_t)p * CC);
#pragma unroll 6
  for (int i = 0; i < 24; ++i) {
    float4 t = v4[i];
    uint2 pk;
    pk.x = (unsigned)f2bf(t.x * r) | ((unsigned)f2bf(t.y * r) << 16);
    pk.y = (unsigned)f2bf(t.z * r) | ((unsigned)f2bf(t.w * r) << 16);
    o[i] = pk;
  }
}

// ---------------- K2: bf16 WMMA Gram + fused top-K -> nn[b][n][9] ----------------
__global__ __launch_bounds__(128) void k_knn(
    const unsigned short* __restrict__ xn, const float* __restrict__ sq,
    int* __restrict__ nnidx) {
  __shared__ float s_score[4][16][16];
  __shared__ float s_mv[4][16][2][KNN];
  __shared__ int   s_mi[4][16][2][KNN];
  const int wid  = threadIdx.x >> 5;
  const int lane = threadIdx.x & 31;
  const int half = lane >> 4, l16 = lane & 15;
  const int rt = (blockIdx.x % NT_BLK) * 4 + wid;
  const int b  = blockIdx.x / NT_BLK;
  const int r0 = rt * 16;
  const int row = r0 + l16;

  // A fragments (rows x K=96, bf16) from [n][c] layout: contiguous 16B chunks
  BF16Frag af[3];
  const unsigned short* xr = xn + ((size_t)b * NN + row) * CC;
#pragma unroll
  for (int f = 0; f < 3; ++f) {
    int kb = 32 * f + (half ? 8 : 0);
    af[f].q[0] = *(const uint4*)(xr + kb);
    af[f].q[1] = *(const uint4*)(xr + kb + 16);
  }
  float sqv[8];
#pragma unroll
  for (int v = 0; v < 8; ++v) sqv[v] = sq[(size_t)b * NN + r0 + v + half * 8];

  float kv[KNN]; int ki[KNN];
#pragma unroll
  for (int t = 0; t < KNN; ++t) { kv[t] = -3.0e38f; ki[t] = 0; }
  float kmin = kv[KNN - 1];

  for (int ct = 0; ct < NT_PER_B; ++ct) {
    int c0 = ct * 16;
    int col = c0 + l16;
    const unsigned short* xc = xn + ((size_t)b * NN + col) * CC;
    BF16Frag bf[3];
#pragma unroll
    for (int f = 0; f < 3; ++f) {
      int kb = 32 * f + (half ? 8 : 0);
      bf[f].q[0] = *(const uint4*)(xc + kb);
      bf[f].q[1] = *(const uint4*)(xc + kb + 16);
    }
    float sqc = sq[(size_t)b * NN + col];
    v8f acc = {0.f,0.f,0.f,0.f,0.f,0.f,0.f,0.f};
    acc = wmma_bf16(af[0].v, bf[0].v, acc);
    acc = wmma_bf16(af[1].v, bf[1].v, acc);
    acc = wmma_bf16(af[2].v, bf[2].v, acc);
#pragma unroll
    for (int v = 0; v < 8; ++v)
      s_score[wid][v + half * 8][l16] = 2.f * acc[v] - sqv[v] - sqc;
    lds_wait();   // intra-wave LDS exchange: wait all lanes' DS stores
#pragma unroll
    for (int j = 0; j < 8; ++j) {
      int cl = half * 8 + j;
      float val = s_score[wid][l16][cl];
      int idx = c0 + cl;
      if (val > kmin) {
        float cv = val; int ci = idx;
#pragma unroll
        for (int t = 0; t < KNN; ++t) {
          bool g = cv > kv[t];
          float tv = kv[t]; int ti = ki[t];
          kv[t] = g ? cv : tv; ki[t] = g ? ci : ti;
          cv = g ? tv : cv;    ci = g ? ti : ci;
        }
        kmin = kv[KNN - 1];
      }
    }
    lds_wait();   // reads complete before next tile's stores (paranoia; same-wave LDS is in-order)
  }
  // merge the two half-lane lists per row (tie-break: lower index)
#pragma unroll
  for (int t = 0; t < KNN; ++t) {
    s_mv[wid][l16][half][t] = kv[t];
    s_mi[wid][l16][half][t] = ki[t];
  }
  lds_wait();
  if (half == 0) {
    int ia = 0, ib = 0;
    int* out = nnidx + ((size_t)b * NN + r0 + l16) * KNN;
    for (int t = 0; t < KNN; ++t) {
      float va = s_mv[wid][l16][0][ia], vb = s_mv[wid][l16][1][ib];
      int   xa = s_mi[wid][l16][0][ia], xb = s_mi[wid][l16][1][ib];
      bool takeA = (va > vb) || (va == vb && xa < xb);
      out[t] = takeA ? xa : xb;
      if (takeA) ++ia; else ++ib;
    }
  }
}

// ---------------- K3: grouped edge conv + bn + relu + max_k -> ym[b][n][192] ----------------
__global__ __launch_bounds__(128) void k_edge(
    const float* __restrict__ xf, const int* __restrict__ nnidx,
    const float* __restrict__ gw, const float* __restrict__ gb,
    const float* __restrict__ bn, float* __restrict__ ym) {
  __shared__ float s_w[4 * 48 * 48];   // 36 KB: full gc_w
  __shared__ float s_sc[C2], s_sh[C2];
  for (int i = threadIdx.x; i < 4 * 48 * 48; i += blockDim.x) s_w[i] = gw[i];
  for (int c = threadIdx.x; c < C2; c += blockDim.x) {
    float g = bn[c], be = bn[C2 + c], mu = bn[2 * C2 + c], va = bn[3 * C2 + c];
    float inv = g / sqrtf(va + 1e-5f);
    s_sc[c] = inv;
    s_sh[c] = gb[c] * inv + be - mu * inv;
  }
  __syncthreads();

  const int wid  = threadIdx.x >> 5;
  const int lane = threadIdx.x & 31;
  const int half = lane >> 4, l16 = lane & 15;
  const int nt = (blockIdx.x % NT_BLK) * 4 + wid;
  const int b  = blockIdx.x / NT_BLK;
  const int n0 = nt * 16, n = n0 + l16;
  const int koff = half ? 2 : 0;
  const float* xi = xf + ((size_t)b * NN + n) * CC;

  // groups 0,1: x_i broadcast over k -> k-independent, no max loop needed
#pragma unroll
  for (int g = 0; g < 2; ++g) {
    v2f a[12];
#pragma unroll
    for (int s = 0; s < 12; ++s) {
      int c = g * 48 + 4 * s + koff;
      a[s].x = xi[c]; a[s].y = xi[c + 1];
    }
    for (int ot = 0; ot < 3; ++ot) {
      int ol = ot * 16 + l16;
      v8f acc = {0.f,0.f,0.f,0.f,0.f,0.f,0.f,0.f};
#pragma unroll
      for (int s = 0; s < 12; ++s) {
        int c = 4 * s + koff;
        v2f bf; bf.x = s_w[(g * 48 + ol) * 48 + c]; bf.y = s_w[(g * 48 + ol) * 48 + c + 1];
        acc = wmma_f32(a[s], bf, acc);
      }
      int og = g * 48 + ol;
      float sc = s_sc[og], sh = s_sh[og];
#pragma unroll
      for (int v = 0; v < 8; ++v) {
        int nr = n0 + v + half * 8;
        ym[((size_t)b * NN + nr) * C2 + og] = fmaxf(acc[v] * sc + sh, 0.f);
      }
    }
  }
  // groups 2,3: (x_j - x_i), max over 9 neighbors
  const int* nrow = nnidx + ((size_t)b * NN + n) * KNN;
#pragma unroll
  for (int g2 = 0; g2 < 2; ++g2) {
    float vmax[3][8];
#pragma unroll
    for (int ot = 0; ot < 3; ++ot)
#pragma unroll
      for (int v = 0; v < 8; ++v) vmax[ot][v] = -3.0e38f;
    for (int k = 0; k < KNN; ++k) {
      int j = nrow[k];
      const float* xj = xf + ((size_t)b * NN + j) * CC;
      v2f a[12];
#pragma unroll
      for (int s = 0; s < 12; ++s) {
        int c = g2 * 48 + 4 * s + koff;
        a[s].x = xj[c] - xi[c];
        a[s].y = xj[c + 1] - xi[c + 1];
      }
#pragma unroll
      for (int ot = 0; ot < 3; ++ot) {
        int ol = ot * 16 + l16;
        v8f acc = {0.f,0.f,0.f,0.f,0.f,0.f,0.f,0.f};
#pragma unroll
        for (int s = 0; s < 12; ++s) {
          int c = 4 * s + koff;
          v2f bf;
          bf.x = s_w[((2 + g2) * 48 + ol) * 48 + c];
          bf.y = s_w[((2 + g2) * 48 + ol) * 48 + c + 1];
          acc = wmma_f32(a[s], bf, acc);
        }
        int og = 96 + g2 * 48 + ol;
        float sc = s_sc[og], sh = s_sh[og];
#pragma unroll
        for (int v = 0; v < 8; ++v)
          vmax[ot][v] = fmaxf(vmax[ot][v], fmaxf(acc[v] * sc + sh, 0.f));
      }
    }
#pragma unroll
    for (int ot = 0; ot < 3; ++ot) {
      int og = 96 + g2 * 48 + ot * 16 + l16;
#pragma unroll
      for (int v = 0; v < 8; ++v) {
        int nr = n0 + v + half * 8;
        ym[((size_t)b * NN + nr) * C2 + og] = vmax[ot][v];
      }
    }
  }
}

// ---------------- K4: fc2 + bn2 + residual -> out[b][c][n] ----------------
__global__ __launch_bounds__(128) void k_fc2(
    const float* __restrict__ ym, const float* __restrict__ w,
    const float* __restrict__ bias, const float* __restrict__ bn,
    const float* __restrict__ xres, float* __restrict__ out) {
  const int wid  = threadIdx.x >> 5;
  const int lane = threadIdx.x & 31;
  const int half = lane >> 4, l16 = lane & 15;
  const int nt = (blockIdx.x % NT_BLK) * 4 + wid;
  const int b  = blockIdx.x / NT_BLK;
  const int n0 = nt * 16, n = n0 + l16;
  const int koff = half ? 2 : 0;

  v2f a[48];
  const float* yb = ym + ((size_t)b * NN + n) * C2;
#pragma unroll
  for (int s = 0; s < 48; ++s) {
    int c = 4 * s + koff;
    a[s].x = yb[c]; a[s].y = yb[c + 1];
  }
  for (int ot = 0; ot < 6; ++ot) {
    int o = ot * 16 + l16;
    v8f acc = {0.f,0.f,0.f,0.f,0.f,0.f,0.f,0.f};
#pragma unroll
    for (int s = 0; s < 48; ++s) {
      int c = 4 * s + koff;
      v2f bf; bf.x = w[o * C2 + c]; bf.y = w[o * C2 + c + 1];
      acc = wmma_f32(a[s], bf, acc);
    }
    float g = bn[o], be = bn[CC + o], mu = bn[2 * CC + o], va = bn[3 * CC + o];
    float inv = g / sqrtf(va + 1e-5f);
    float sh  = bias[o] * inv + be - mu * inv;
#pragma unroll
    for (int v = 0; v < 8; ++v) {
      int nr = n0 + v + half * 8;
      size_t oi = ((size_t)b * CC + o) * NN + nr;
      out[oi] = acc[v] * inv + sh + xres[oi];
    }
  }
}

// ---------------- workspace layout (bytes) ----------------
#define XF_OFF 0u                       // 4*3136*96*4   = 4,816,896
#define XN_OFF 4816896u                 // 4*3136*96*2   = 2,408,448
#define SQ_OFF 7225344u                 // 4*3136*4      =    50,176
#define NN_OFF 7275520u                 // 4*3136*9*4    =   451,584
#define YM_OFF 7727104u                 // 4*3136*192*4  = 9,633,792  (total ~17.4 MB)

extern "C" void kernel_launch(void* const* d_in, const int* in_sizes, int n_in,
                              void* d_out, int out_size, void* d_ws, size_t ws_size,
                              hipStream_t stream) {
  const float* x     = (const float*)d_in[0];
  const float* fc1_w = (const float*)d_in[1];
  const float* fc1_b = (const float*)d_in[2];
  const float* bn1   = (const float*)d_in[3];
  const float* gc_w  = (const float*)d_in[4];
  const float* gc_b  = (const float*)d_in[5];
  const float* bng   = (const float*)d_in[6];
  const float* fc2_w = (const float*)d_in[7];
  const float* fc2_b = (const float*)d_in[8];
  const float* bn2   = (const float*)d_in[9];
  float* out = (float*)d_out;

  char* ws = (char*)d_ws;
  float*          xf = (float*)(ws + XF_OFF);
  unsigned short* xb = (unsigned short*)(ws + XN_OFF);
  float*          sq = (float*)(ws + SQ_OFF);
  int*            nn = (int*)(ws + NN_OFF);
  float*          ym = (float*)(ws + YM_OFF);

  dim3 blk(128);
  dim3 grid(BB * NT_BLK);
  k_fc1 <<<grid, blk, 0, stream>>>(x, fc1_w, fc1_b, bn1, xf);
  k_norm<<<dim3((BB * NN + 255) / 256), dim3(256), 0, stream>>>(xf, xb, sq);
  k_knn <<<grid, blk, 0, stream>>>(xb, sq, nn);
  k_edge<<<grid, blk, 0, stream>>>(xf, nn, gc_w, gc_b, bng, ym);
  k_fc2 <<<grid, blk, 0, stream>>>(ym, fc2_w, fc2_b, bn2, x, out);
}
